// verifyZ_32504312496837
// MI455X (gfx1250) — matrix operand order repaired and checked by reading, not compiled, not run on hardware
//
#include <hip/hip_runtime.h>
#include <hip/hip_bf16.h>
#include <stdint.h>

#define N_NODES  4096
#define N_EDGES  16384
#define N_GRAPHS 128
#define NWORDS   128   // 4096 bits / 32 per node bitset

typedef __attribute__((ext_vector_type(16))) _Float16 v16h;
typedef __attribute__((ext_vector_type(8)))  float    v8f;

// ---------------------------------------------------------------------------
// Pad/convert both feature matrices to dense items x 16 f16 tiles:
// cols 0..nfeat-1 = data, col nfeat = 1.0 (count column), rest = 0.
// Removes all divergence from the WMMA inner loop's B construction.
// ---------------------------------------------------------------------------
__global__ void pad_f16_kernel(const float* __restrict__ x,
                               const float* __restrict__ ea,
                               _Float16* __restrict__ xpad,
                               _Float16* __restrict__ epad) {
  int idx = blockIdx.x * blockDim.x + threadIdx.x;
  if (idx < N_NODES * 16) {
    int item = idx >> 4, col = idx & 15;
    float v = (col < 13) ? x[item * 13 + col] : (col == 13 ? 1.0f : 0.0f);
    xpad[idx] = (_Float16)v;
  } else {
    int j = idx - N_NODES * 16;
    if (j < N_EDGES * 16) {
      int item = j >> 4, col = j & 15;
      float v = (col < 4) ? ea[item * 4 + col] : (col == 4 ? 1.0f : 0.0f);
      epad[j] = (_Float16)v;
    }
  }
}

// ---------------------------------------------------------------------------
// Segment-sum as one-hot matmul on the matrix engine.
// Blocks 0..7  : node features  (16-graph tile each, K = 4096 items)
// Blocks 8..15 : edge features  (16-graph tile each, K = 16384 items)
// One wave per tile; K unrolled x2 -> two v_wmma per iteration chained on C.
// Layouts used here (wave32; C/D as ISA 7.12.2, A/B contiguous k, not ISA order):
//   A  : lanes0-15 M=l15, elem e -> K = e
//        lanes16-31 M=l15, elem e -> K = e+16
//   B  : N = l15, elem e -> K = e + 16*half
//   C/D: VGPR r -> row gbase + r + 8*half, col l15
// Membership one-hot and the count column are exact in f16; C/D accumulate f32.
// ---------------------------------------------------------------------------
__global__ __launch_bounds__(32) void wmma_seg_sum_kernel(
    const _Float16* __restrict__ xpad, const int* __restrict__ xseg,
    const _Float16* __restrict__ epad, const int* __restrict__ eseg,
    float* __restrict__ xsum, float* __restrict__ esum) {
  int blk = blockIdx.x;
  const _Float16* pad;
  const int* seg;
  float* out;
  int n_items, gt;
  if (blk < 8) { pad = xpad; seg = xseg; out = xsum; n_items = N_NODES;  gt = blk; }
  else         { pad = epad; seg = eseg; out = esum; n_items = N_EDGES;  gt = blk - 8; }

  const int lane  = threadIdx.x & 31;
  const int half  = lane >> 4;
  const int l15   = lane & 15;
  const int gbase = gt * 16;
  const int gid   = gbase + l15;

  int kmapA[16];
#pragma unroll
  for (int e = 0; e < 16; ++e)
    kmapA[e] = half * 16 + e;

  v8f c = {0.f, 0.f, 0.f, 0.f, 0.f, 0.f, 0.f, 0.f};

  for (int kb = 0; kb < n_items; kb += 64) {
    v16h a0, b0, a1, b1;
#pragma unroll
    for (int e = 0; e < 16; ++e) {
      a0[e] = (seg[kb      + kmapA[e]] == gid) ? (_Float16)1.0f : (_Float16)0.0f;
      a1[e] = (seg[kb + 32 + kmapA[e]] == gid) ? (_Float16)1.0f : (_Float16)0.0f;
    }
    const _Float16* prow0 = pad + (size_t)(kb + half * 16) * 16 + l15;
    const _Float16* prow1 = prow0 + 32 * 16;
#pragma unroll
    for (int e = 0; e < 16; ++e) {
      b0[e] = prow0[e * 16];
      b1[e] = prow1[e * 16];
    }
    c = __builtin_amdgcn_wmma_f32_16x16x32_f16(false, a0, false, b0,
                                               (short)0, c, false, false);
    c = __builtin_amdgcn_wmma_f32_16x16x32_f16(false, a1, false, b1,
                                               (short)0, c, false, false);
  }
#pragma unroll
  for (int r = 0; r < 8; ++r)
    out[(gbase + r + half * 8) * 16 + l15] = c[r];
}

// Scatter edge_attr into dst nodes + build direct-in-edge adjacency bitset.
__global__ void scatter_edges_kernel(const float* __restrict__ ea,
                                     const int* __restrict__ src,
                                     const int* __restrict__ dst,
                                     float* __restrict__ edge_input,
                                     unsigned* __restrict__ Abits) {
  int e = blockIdx.x * blockDim.x + threadIdx.x;
  if (e >= N_EDGES) return;
  int d = dst[e], s = src[e];
  float4 v = ((const float4*)ea)[e];
  atomicAdd(&edge_input[d * 4 + 0], v.x);
  atomicAdd(&edge_input[d * 4 + 1], v.y);
  atomicAdd(&edge_input[d * 4 + 2], v.z);
  atomicAdd(&edge_input[d * 4 + 3], v.w);
  atomicOr(&Abits[(size_t)d * NWORDS + (s >> 5)], 1u << (s & 31));
}

// Per-node flow terms + segment max (clamped at 0 -> nonneg floats order as
// uints, init 0), plus aroma flags and per-graph aroma counts.
__global__ void node_flow_kernel(const float* __restrict__ x,
                                 const float* __restrict__ edge_input,
                                 const int* __restrict__ batch,
                                 unsigned* __restrict__ maxflow /*128x4*/,
                                 unsigned* __restrict__ aroma,
                                 unsigned* __restrict__ num_aroma) {
  int i = blockIdx.x * blockDim.x + threadIdx.x;
  if (i >= N_NODES) return;
  const float* xi = x + (size_t)i * 13;
  float4 ei = ((const float4*)edge_input)[i];
  float in_x = ei.x * 1.0f + ei.y * 2.0f + ei.z * 3.0f + ei.w * 1.5f;
  float xv = xi[0]*4.f + xi[1]*3.f + xi[2]*2.f + xi[3]*1.f + xi[4]*6.f
           + xi[5]*2.f + xi[6]*1.f + xi[7]*1.f + xi[8]*1.f
           + xi[10]*1.f - xi[11];                 // W9 then W3=[0,1,-1]
  float x2   = xi[12];
  float inx2 = (ei.w > 0.5f) ? 1.0f : 0.0f;
  float c0 = in_x - xv, c1 = inx2 - x2, c2 = x2 - inx2;
  int g = batch[i];
  atomicMax(&maxflow[g * 4 + 0], __float_as_uint(fmaxf(c0, 0.0f)));
  atomicMax(&maxflow[g * 4 + 1], __float_as_uint(fmaxf(c1, 0.0f)));
  atomicMax(&maxflow[g * 4 + 2], __float_as_uint(fmaxf(c2, 0.0f)));
  unsigned ar = (x2 > 0.5f) ? 1u : 0u;
  aroma[i] = ar;
  if (ar) atomicAdd(&num_aroma[g], 1u);
}

// u1 = diag(aroma) | A.diag(aroma): seed diagonal bit + 1-hop bits from
// aroma sources (u0 is diagonal, so the first sim_pass is edge-sparse).
__global__ void seed_u1_kernel(const int* __restrict__ src,
                               const int* __restrict__ dst,
                               const unsigned* __restrict__ aroma,
                               unsigned* __restrict__ u1) {
  int t = blockIdx.x * blockDim.x + threadIdx.x;
  if (t < N_NODES) {
    if (aroma[t])
      atomicOr(&u1[(size_t)t * NWORDS + (t >> 5)], 1u << (t & 31));
  } else if (t < N_NODES + N_EDGES) {
    int e = t - N_NODES;
    int s = src[e];
    if (aroma[s])
      atomicOr(&u1[(size_t)dst[e] * NWORDS + (s >> 5)], 1u << (s & 31));
  }
}

// uout[dst] |= uin[src] over all edges; one thread per (edge, bitset word).
__global__ void prop_or_kernel(const int* __restrict__ src,
                               const int* __restrict__ dst,
                               const unsigned* __restrict__ uin,
                               unsigned* __restrict__ uout) {
  int idx = blockIdx.x * blockDim.x + threadIdx.x;
  int e = idx >> 7;           // NWORDS == 128
  int w = idx & (NWORDS - 1);
  if (e >= N_EDGES) return;
  unsigned v = uin[(size_t)src[e] * NWORDS + w];
  if (v) atomicOr(&uout[(size_t)dst[e] * NWORDS + w], v);
}

// out_x[i] > 0  <=>  (F[i] & ~A[i] & ~bit_i) != 0 ; count per graph.
__global__ void cycle_kernel(const unsigned* __restrict__ F,
                             const unsigned* __restrict__ A,
                             const int* __restrict__ batch,
                             unsigned* __restrict__ num_cycle) {
  int i = blockIdx.x * blockDim.x + threadIdx.x;
  if (i >= N_NODES) return;
  const unsigned* Fi = F + (size_t)i * NWORDS;
  const unsigned* Ai = A + (size_t)i * NWORDS;
  __builtin_prefetch(Ai, 0, 1);
  int sw = i >> 5;
  unsigned sb = 1u << (i & 31);
  bool nz = false;
  for (int w = 0; w < NWORDS; ++w) {
    unsigned m = Fi[w] & ~Ai[w];
    if (w == sw) m &= ~sb;
    if (m) { nz = true; break; }
  }
  if (nz) atomicAdd(&num_cycle[batch[i]], 1u);
}

__global__ void finalize_kernel(const float* __restrict__ xsum,
                                const float* __restrict__ esum,
                                const unsigned* __restrict__ maxflow,
                                const unsigned* __restrict__ num_aroma,
                                const unsigned* __restrict__ num_cycle,
                                float* __restrict__ out) {
  int g = blockIdx.x * blockDim.x + threadIdx.x;
  if (g >= N_GRAPHS) return;
  // loss1: node-feature segment mean vs 0.1, coef 3
  float cnt = fmaxf(xsum[g * 16 + 13], 1.0f);
  float l1 = 0.f;
  for (int f = 0; f < 13; ++f) {
    float d = xsum[g * 16 + f] / cnt - 0.1f;
    l1 += d * d;
  }
  l1 *= 3.0f;
  // loss2: edge-feature segment mean vs 0.25, coef 3
  float ecraw = esum[g * 16 + 4];
  float ecnt = fmaxf(ecraw, 1.0f);
  float l2 = 0.f;
  for (int f = 0; f < 4; ++f) {
    float d = esum[g * 16 + f] / ecnt - 0.25f;
    l2 += d * d;
  }
  l2 *= 3.0f;
  // loss3: leaky_relu(edge_count - 21, 0.3)^2, coef 1
  float a3 = ecraw - 21.0f;
  float lr3 = (a3 >= 0.f) ? a3 : 0.3f * a3;
  float l3 = lr3 * lr3;
  // loss4: sum of clamped segment maxes, coef 4
  float l4 = (__uint_as_float(maxflow[g * 4 + 0]) +
              __uint_as_float(maxflow[g * 4 + 1]) +
              __uint_as_float(maxflow[g * 4 + 2])) * 4.0f;
  // loss5: leaky_relu(num_cycle - num_aroma, 0.02)^2, coef 3
  float d5 = (float)num_cycle[g] - (float)num_aroma[g];
  float lr5 = (d5 >= 0.f) ? d5 : 0.02f * d5;
  float l5 = lr5 * lr5 * 3.0f;
  out[g] = -(l1 + l2 + l3 + l4 + l5);
}

extern "C" void kernel_launch(void* const* d_in, const int* in_sizes, int n_in,
                              void* d_out, int out_size, void* d_ws, size_t ws_size,
                              hipStream_t stream) {
  const float* x      = (const float*)d_in[0];   // 4096 x 13
  const float* ea     = (const float*)d_in[1];   // 16384 x 4
  const int*   batch  = (const int*)  d_in[2];   // 4096
  const int*   eidx   = (const int*)  d_in[3];   // 2 x 16384
  const int*   ebatch = (const int*)  d_in[4];   // 16384
  const int*   src = eidx;
  const int*   dst = eidx + N_EDGES;
  float* out = (float*)d_out;

  uint8_t* p = (uint8_t*)d_ws;
  float*    xsum       = (float*)p;    p += (size_t)N_GRAPHS * 16 * sizeof(float);
  float*    esum       = (float*)p;    p += (size_t)N_GRAPHS * 16 * sizeof(float);
  float*    edge_input = (float*)p;    p += (size_t)N_NODES * 4 * sizeof(float);
  unsigned* maxflow    = (unsigned*)p; p += (size_t)N_GRAPHS * 4 * sizeof(unsigned);
  unsigned* num_aroma  = (unsigned*)p; p += (size_t)N_GRAPHS * sizeof(unsigned);
  unsigned* num_cycle  = (unsigned*)p; p += (size_t)N_GRAPHS * sizeof(unsigned);
  unsigned* aroma      = (unsigned*)p; p += (size_t)N_NODES * sizeof(unsigned);
  const size_t bits_bytes = (size_t)N_NODES * NWORDS * sizeof(unsigned); // 2 MB
  unsigned*  Abits = (unsigned*)p;  p += bits_bytes;
  unsigned*  u1    = (unsigned*)p;  p += bits_bytes;
  unsigned*  u2    = (unsigned*)p;  p += bits_bytes;
  unsigned*  Fb    = (unsigned*)p;  p += bits_bytes;
  _Float16*  xpad  = (_Float16*)p;  p += (size_t)N_NODES * 16 * sizeof(_Float16);
  _Float16*  epad  = (_Float16*)p;  p += (size_t)N_EDGES * 16 * sizeof(_Float16);
  size_t used = (size_t)(p - (uint8_t*)d_ws);

  // Zero all accumulators / bitsets (graph-capture safe).
  hipMemsetAsync(d_ws, 0, used, stream);

  // Pad + f16-convert both feature matrices (count column baked in).
  pad_f16_kernel<<<(N_NODES * 16 + N_EDGES * 16 + 255) / 256, 256, 0, stream>>>(
      x, ea, xpad, epad);

  // Both segment-sum matmuls in one launch: 16 tiles of 16 graphs, 1 wave each.
  wmma_seg_sum_kernel<<<16, 32, 0, stream>>>(xpad, batch, epad, ebatch,
                                             xsum, esum);

  // Edge scatter: edge_input (f32 adds) + adjacency bitset.
  scatter_edges_kernel<<<(N_EDGES + 255) / 256, 256, 0, stream>>>(
      ea, src, dst, edge_input, Abits);

  // Per-node flow terms, segment max, aroma flags/counts.
  node_flow_kernel<<<(N_NODES + 255) / 256, 256, 0, stream>>>(
      x, edge_input, batch, maxflow, aroma, num_aroma);

  // Reachability pass 1 (u0 diagonal -> edge-sparse seed of u1).
  seed_u1_kernel<<<(N_NODES + N_EDGES + 255) / 256, 256, 0, stream>>>(
      src, dst, aroma, u1);

  // Pass 2: u2 = u1 | A.u1 (copy then OR-propagate).
  hipMemcpyAsync(u2, u1, bits_bytes, hipMemcpyDeviceToDevice, stream);
  prop_or_kernel<<<(N_EDGES * NWORDS) / 256, 256, 0, stream>>>(src, dst, u1, u2);

  // F = A.u2 (boolean).
  prop_or_kernel<<<(N_EDGES * NWORDS) / 256, 256, 0, stream>>>(src, dst, u2, Fb);

  // Cycle nodes per graph.
  cycle_kernel<<<(N_NODES + 255) / 256, 256, 0, stream>>>(Fb, Abits, batch, num_cycle);

  // Combine everything into the 128 outputs.
  finalize_kernel<<<1, 128, 0, stream>>>(xsum, esum, maxflow, num_aroma,
                                         num_cycle, out);
}
